// FoundationalTimeSeriesModel_14267881357516
// MI455X (gfx1250) — compile-verified, matrix operand order checked by reading, not run on hardware
//
#include <hip/hip_runtime.h>
#include <math.h>

// ---------------- Types for CDNA5 WMMA ----------------
typedef __attribute__((ext_vector_type(16))) __bf16 v16bf;
typedef __attribute__((ext_vector_type(8)))  __bf16 v8bf;    // 16 bytes
typedef __attribute__((ext_vector_type(8)))  float  v8f;
typedef __attribute__((ext_vector_type(4)))  float  v4f;
typedef __attribute__((ext_vector_type(4)))  unsigned v4u;

// Model dims
#define BB    16
#define LL    512
#define SS    32
#define NSEQ  512      // B*S
#define PROJC 64
#define TCNC  128
#define DM    256
#define NHEAD 8
#define HD    32
#define FFD   512
#define NE    16
#define EHID  512
#define EOUT  256

// GEMM flags
#define GF_RELU 1
#define GF_ADD  2
#define GF_BIAS 4
// Conv flags
#define CF_RELU  1
#define CF_RESID 2

#define SHUF16(a, b) __builtin_shufflevector(a, b, 0,1,2,3,4,5,6,7,8,9,10,11,12,13,14,15)

// ---------------- weight repack kernels (run every call; cheap) ----------------
// fp32 (K,N) row-major -> bf16 (N,K) row-major (B^T), batched via grid.z
__global__ void repack_wt_kernel(const float* __restrict__ W, __bf16* __restrict__ O,
                                 int K, int N) {
  size_t base = (size_t)blockIdx.z * K * N;
  int t = blockIdx.x * blockDim.x + threadIdx.x;
  if (t >= K * N) return;
  int k = t / N, n = t % N;
  O[base + (size_t)n * K + k] = (__bf16)W[base + t];
}

// fp32 conv weight (cout,cin,KT) -> bf16 [KT][cout][cin]
__global__ void repack_cw_kernel(const float* __restrict__ W, __bf16* __restrict__ O,
                                 int cout, int cin, int KT) {
  int t = blockIdx.x * blockDim.x + threadIdx.x;
  if (t >= cout * cin * KT) return;
  int co = t / (cin * KT);
  int r = t % (cin * KT);
  int ci = r / KT, kt = r % KT;
  O[((size_t)kt * cout + co) * cin + ci] = (__bf16)W[t];
}

// ---------------- Kernel 1: projection + positional encoding ----------------
// x:(B,L,S) -> X0 bf16 (NSEQ, LL, PROJC) channel-contiguous;  last_val[n]=x[b,L-1,s]
__global__ void proj_pe_kernel(const float* __restrict__ x,
                               const float* __restrict__ pw,
                               const float* __restrict__ pb,
                               __bf16* __restrict__ X0,
                               float* __restrict__ last_val) {
  int idx = blockIdx.x;          // n*LL + l
  int c = threadIdx.x;           // 0..63
  int n = idx >> 9;
  int l = idx & 511;
  int b = n >> 5, s = n & 31;
  float xv = x[((size_t)b * LL + l) * SS + s];
  int i = c >> 1;
  float div = expf((float)(2 * i) * (-9.210340372f / 64.f));
  float ang = (float)l * div;
  float pe = (c & 1) ? cosf(ang) : sinf(ang);
  float h = xv * pw[c] + pb[c] + pe;
  X0[((size_t)n * LL + l) * PROJC + c] = (__bf16)h;
  if (c == 0 && l == LL - 1) last_val[n] = xv;
}

// ---------------- Kernel 2: dilated causal conv as WMMA GEMM ----------------
// X:(NSEQ,LL,cin) bf16 -> Y:(NSEQ,LL,cout) bf16.  CW: bf16 [KT][cout][cin].
// 2x2 tiling per wave: 2 cout-tiles x 2 timestep-tiles = 4 independent WMMA chains
// (A fragments shared across l-tiles, B fragments shared across cout-tiles).
__global__ void conv_gemm_kernel(const __bf16* __restrict__ X,
                                 const __bf16* __restrict__ CW,
                                 const float* __restrict__ bias,
                                 const __bf16* __restrict__ R,
                                 __bf16* __restrict__ Y,
                                 int cin, int cout, int dil, int KT, int flags) {
  const int tilesC2 = cout >> 5, tilesL2 = LL >> 5;
  int tile = blockIdx.x * (blockDim.x >> 5) + (threadIdx.x >> 5);
  int nTiles = NSEQ * tilesC2 * tilesL2;
  if (tile >= nTiles) return;                   // wave-uniform: EXEC stays all-ones for WMMA
  int seq = tile / (tilesC2 * tilesL2);
  int rem = tile % (tilesC2 * tilesL2);
  int co0 = (rem / tilesL2) << 5;               // 32 channels per wave
  int l0  = (rem % tilesL2) << 5;               // 32 timesteps per wave
  int lane = threadIdx.x & 31, half = lane >> 4, lm = lane & 15;
  const __bf16* Xs = X + (size_t)seq * LL * cin;

  v8f acc00, acc10, acc01, acc11;               // acc[coTile][lTile]
#pragma unroll
  for (int i = 0; i < 8; ++i) { acc00[i] = 0.f; acc10[i] = 0.f; acc01[i] = 0.f; acc11[i] = 0.f; }

  for (int kt = 0; kt < KT; ++kt) {
    int shift = (KT - 1 - kt) * dil;
    int lcolA = l0 + lm - shift;
    int lcolB = l0 + 16 + lm - shift;
    int lsafeA = lcolA < 0 ? 0 : lcolA;
    int lsafeB = lcolB < 0 ? 0 : lcolB;
    v4u mskA = (v4u)(lcolA < 0 ? 0u : 0xffffffffu);   // branch-free causal zeroing
    v4u mskB = (v4u)(lcolB < 0 ? 0u : 0xffffffffu);
    const __bf16* XcolA = Xs + (size_t)lsafeA * cin + (half << 3);
    const __bf16* XcolB = Xs + (size_t)lsafeB * cin + (half << 3);
    const __bf16* Wk0 = CW + (size_t)kt * cout * cin + (size_t)(co0 + lm) * cin + (half << 3);
    const __bf16* Wk1 = Wk0 + (size_t)16 * cin;
    for (int c0 = 0; c0 < cin; c0 += 32) {
      __builtin_prefetch(XcolA + c0 + 64, 0, 1);
      v4u a0u = *(const v4u*)(XcolA + c0) & mskA;
      v4u a1u = *(const v4u*)(XcolA + c0 + 16) & mskA;
      v4u b0u = *(const v4u*)(XcolB + c0) & mskB;
      v4u b1u = *(const v4u*)(XcolB + c0 + 16) & mskB;
      v16bf bfA = SHUF16(__builtin_bit_cast(v8bf, a0u), __builtin_bit_cast(v8bf, a1u));
      v16bf bfB = SHUF16(__builtin_bit_cast(v8bf, b0u), __builtin_bit_cast(v8bf, b1u));
      v16bf af0 = SHUF16(*(const v8bf*)(Wk0 + c0), *(const v8bf*)(Wk0 + c0 + 16));
      v16bf af1 = SHUF16(*(const v8bf*)(Wk1 + c0), *(const v8bf*)(Wk1 + c0 + 16));
      acc00 = __builtin_amdgcn_wmma_f32_16x16x32_bf16(false, af0, false, bfA,
                                                      (short)0, acc00, false, false);
      acc10 = __builtin_amdgcn_wmma_f32_16x16x32_bf16(false, af1, false, bfA,
                                                      (short)0, acc10, false, false);
      acc01 = __builtin_amdgcn_wmma_f32_16x16x32_bf16(false, af0, false, bfB,
                                                      (short)0, acc01, false, false);
      acc11 = __builtin_amdgcn_wmma_f32_16x16x32_bf16(false, af1, false, bfB,
                                                      (short)0, acc11, false, false);
    }
  }
  // epilogue: per l-tile, lane owns 8-channel runs at [co0+8*half] and [co0+16+8*half]
#pragma unroll
  for (int lt = 0; lt < 2; ++lt) {
    int l = l0 + (lt << 4) + lm;
    size_t rowbase = ((size_t)seq * LL + l) * cout;
#pragma unroll
    for (int t2 = 0; t2 < 2; ++t2) {
      int ch0 = co0 + (t2 << 4) + (half << 3);
      v4f blo = *(const v4f*)(bias + ch0);
      v4f bhi = *(const v4f*)(bias + ch0 + 4);
      v8f* accp = (t2 == 0) ? (lt == 0 ? &acc00 : &acc01)
                            : (lt == 0 ? &acc10 : &acc11);
      v8bf outv;
      if (flags & CF_RESID) {
        v8bf rv = *(const v8bf*)(R + rowbase + ch0);
#pragma unroll
        for (int r = 0; r < 8; ++r) {
          float v = (*accp)[r] + (r < 4 ? blo[r & 3] : bhi[r & 3]) + (float)rv[r];
          if (flags & CF_RELU) v = fmaxf(v, 0.f);
          outv[r] = (__bf16)v;
        }
      } else {
#pragma unroll
        for (int r = 0; r < 8; ++r) {
          float v = (*accp)[r] + (r < 4 ? blo[r & 3] : bhi[r & 3]);
          if (flags & CF_RELU) v = fmaxf(v, 0.f);
          outv[r] = (__bf16)v;
        }
      }
      *(v8bf*)(Y + rowbase + ch0) = outv;
    }
  }
}

// ---------------- Kernel 3: LN over channels + masked mean-pool + last tap ----------------
// H: (NSEQ, LL, TCNC) bf16
__global__ void ln_pool_kernel(const __bf16* __restrict__ H,
                               const float* __restrict__ g, const float* __restrict__ bta,
                               const float* __restrict__ mask,
                               float* __restrict__ pooled, __bf16* __restrict__ pooled_bf,
                               float* __restrict__ tcn_last) {
  int n = blockIdx.x;
  int c = threadIdx.x;               // 128
  __shared__ float red[TCNC];
  float msk = mask[n];
  float acc = 0.f, last = 0.f;
  const __bf16* Hn = H + (size_t)n * LL * TCNC;
  for (int l = 0; l < LL; ++l) {
    float v = (float)Hn[(size_t)l * TCNC + c];
    red[c] = v; __syncthreads();
    for (int off = TCNC >> 1; off; off >>= 1) { if (c < off) red[c] += red[c + off]; __syncthreads(); }
    float mean = red[0] / (float)TCNC; __syncthreads();
    float d = v - mean;
    red[c] = d * d; __syncthreads();
    for (int off = TCNC >> 1; off; off >>= 1) { if (c < off) red[c] += red[c + off]; __syncthreads(); }
    float var = red[0] / (float)TCNC;
    float ln = d * rsqrtf(var + 1e-5f) * g[c] + bta[c];
    acc += ln;
    if (l == LL - 1) last = ln;
    __syncthreads();
  }
  float pv = acc / (float)LL * msk;
  pooled[(size_t)n * TCNC + c] = pv;
  pooled_bf[(size_t)n * TCNC + c] = (__bf16)pv;
  tcn_last[(size_t)n * TCNC + c] = last * msk;
}

// ---------------- Kernel 4: generic batched WMMA GEMM ----------------
// A: bf16 (M,K) row-major.  Bt: bf16 (N,K) row-major (pre-transposed weights).
// Each wave computes TWO adjacent 16-col tiles (shared A fragment, 2 indep WMMAs).
// C: fp32 out; optional bf16 copy Cbf for chained GEMMs.
__global__ void gemm_kernel(const __bf16* __restrict__ A, int lda, unsigned long long asb,
                            const __bf16* __restrict__ Bt, unsigned long long bsb,
                            const float* __restrict__ bias, unsigned long long biassb,
                            float* __restrict__ C, int ldc, unsigned long long csb,
                            __bf16* __restrict__ Cbf,
                            int M, int N, int Kd, int flags) {
  int batch = blockIdx.z;
  A += (size_t)batch * asb; Bt += (size_t)batch * bsb; C += (size_t)batch * csb;
  if (bias) bias += (size_t)batch * biassb;
  if (Cbf) Cbf += (size_t)batch * csb;
  int tilesM = M >> 4, tilesN2 = N >> 5;
  int tile = blockIdx.x * (blockDim.x >> 5) + (threadIdx.x >> 5);
  if (tile >= tilesM * tilesN2) return;
  int m0 = (tile / tilesN2) << 4;
  int n0 = (tile % tilesN2) << 5;               // 32 columns per wave
  int lane = threadIdx.x & 31, half = lane >> 4, lm = lane & 15;

  const __bf16* Ap  = A + (size_t)(m0 + lm) * lda + (half << 3);
  const __bf16* Bp0 = Bt + (size_t)(n0 + lm) * Kd + (half << 3);
  const __bf16* Bp1 = Bp0 + (size_t)16 * Kd;

  v8f acc0, acc1;
#pragma unroll
  for (int i = 0; i < 8; ++i) { acc0[i] = 0.f; acc1[i] = 0.f; }

  for (int k0 = 0; k0 < Kd; k0 += 32) {
    v16bf af = SHUF16(*(const v8bf*)(Ap + k0), *(const v8bf*)(Ap + k0 + 16));
    v16bf bf0 = SHUF16(*(const v8bf*)(Bp0 + k0), *(const v8bf*)(Bp0 + k0 + 16));
    v16bf bf1 = SHUF16(*(const v8bf*)(Bp1 + k0), *(const v8bf*)(Bp1 + k0 + 16));
    acc0 = __builtin_amdgcn_wmma_f32_16x16x32_bf16(false, af, false, bf0,
                                                   (short)0, acc0, false, false);
    acc1 = __builtin_amdgcn_wmma_f32_16x16x32_bf16(false, af, false, bf1,
                                                   (short)0, acc1, false, false);
  }
#pragma unroll
  for (int t2 = 0; t2 < 2; ++t2) {
    int nn = n0 + (t2 << 4) + lm;
    float bv = (flags & GF_BIAS) ? bias[nn] : 0.f;
    v8f* accp = t2 ? &acc1 : &acc0;
#pragma unroll
    for (int r = 0; r < 8; ++r) {
      int m = m0 + r + (half << 3);
      float v = (*accp)[r] + bv;
      if (flags & GF_RELU) v = fmaxf(v, 0.f);
      size_t idx = (size_t)m * ldc + nn;
      if (flags & GF_ADD) v += C[idx];
      C[idx] = v;
      if (Cbf) Cbf[idx] = (__bf16)v;
    }
  }
}

// ---------------- small elementwise / LN / attention / gating ----------------
__global__ void add_pos_kernel(float* __restrict__ z, const float* __restrict__ pos) {
  int n = blockIdx.x, c = threadIdx.x;
  z[(size_t)n * DM + c] += pos[(size_t)(n & 31) * DM + c];
}

__global__ void ln_rows_kernel(const float* __restrict__ in, const float* __restrict__ g,
                               const float* __restrict__ b, float* __restrict__ out,
                               __bf16* __restrict__ obf, int W) {
  extern __shared__ float red[];
  int row = blockIdx.x, c = threadIdx.x;
  float v = in[(size_t)row * W + c];
  red[c] = v; __syncthreads();
  for (int off = W >> 1; off; off >>= 1) { if (c < off) red[c] += red[c + off]; __syncthreads(); }
  float mean = red[0] / (float)W; __syncthreads();
  float d = v - mean;
  red[c] = d * d; __syncthreads();
  for (int off = W >> 1; off; off >>= 1) { if (c < off) red[c] += red[c + off]; __syncthreads(); }
  float var = red[0] / (float)W;
  float o = d * rsqrtf(var + 1e-5f) * g[c] + b[c];
  out[(size_t)row * W + c] = o;
  if (obf) obf[(size_t)row * W + c] = (__bf16)o;
}

__global__ void attn_kernel(const float* __restrict__ q, const float* __restrict__ k,
                            const float* __restrict__ v, const float* __restrict__ mask,
                            float* __restrict__ o, __bf16* __restrict__ obf) {
  int b = blockIdx.x / NHEAD, h = blockIdx.x % NHEAD;
  int lane = threadIdx.x;                 // 32 = one wave, one query row per lane
  __shared__ float ks[SS][HD], vs[SS][HD];
  float qr[HD];
#pragma unroll
  for (int d = 0; d < HD; ++d) {
    size_t base = ((size_t)(b * SS + lane)) * DM + h * HD + d;
    qr[d] = q[base];
    ks[lane][d] = k[base];
    vs[lane][d] = v[base];
  }
  __syncthreads();
  float scr[SS];
  float mx = -1e30f;
  for (int j = 0; j < SS; ++j) {
    float s = 0.f;
#pragma unroll
    for (int d = 0; d < HD; ++d) s += qr[d] * ks[j][d];
    s *= 0.1767766953f;                   // 1/sqrt(32)
    if (mask[b * SS + j] == 0.f) s += -1e9f;
    scr[j] = s;
    mx = fmaxf(mx, s);
  }
  float sum = 0.f;
  for (int j = 0; j < SS; ++j) { scr[j] = expf(scr[j] - mx); sum += scr[j]; }
  float inv = 1.f / sum;
  float orow[HD];
#pragma unroll
  for (int d = 0; d < HD; ++d) orow[d] = 0.f;
  for (int j = 0; j < SS; ++j) {
    float p = scr[j] * inv;
#pragma unroll
    for (int d = 0; d < HD; ++d) orow[d] += p * vs[j][d];
  }
#pragma unroll
  for (int d = 0; d < HD; ++d) {
    size_t idx = ((size_t)(b * SS + lane)) * DM + h * HD + d;
    o[idx] = orow[d];
    obf[idx] = (__bf16)orow[d];
  }
}

__global__ void ctx_stats_kernel(const float* __restrict__ ctx, const float* __restrict__ mask,
                                 float* __restrict__ ctxm, __bf16* __restrict__ ctxm_bf,
                                 float* __restrict__ meanb, __bf16* __restrict__ meanb_bf,
                                 float* __restrict__ stdb, float* __restrict__ rin) {
  int b = blockIdx.x, c = threadIdx.x;    // 256
  float s1 = 0.f, s2 = 0.f, nact = 0.f;
  for (int s = 0; s < SS; ++s) {
    float mk = mask[b * SS + s];
    float vv = ctx[((size_t)(b * SS + s)) * DM + c];
    float cm = vv * mk;
    ctxm[((size_t)(b * SS + s)) * DM + c] = cm;
    ctxm_bf[((size_t)(b * SS + s)) * DM + c] = (__bf16)cm;
    s1 += cm; s2 += cm * cm * mk; nact += mk;
  }
  nact = fmaxf(nact, 1.f);
  float mean = s1 / nact;
  float var = fmaxf(s2 / nact - mean * mean, 1e-6f);
  float sd = sqrtf(var);
  meanb[(size_t)b * DM + c] = mean;
  meanb_bf[(size_t)b * DM + c] = (__bf16)mean;
  stdb[(size_t)b * DM + c]  = sd;
  rin[(size_t)b * (2 * DM) + c]      = mean;
  rin[(size_t)b * (2 * DM) + DM + c] = sd;
}

// gating: logits=gin@gw+gb, softmax, top-2, aux stats into stats[33] via atomics
__global__ void gate_kernel(const float* __restrict__ gin, int G,
                            const float* __restrict__ gw, const float* __restrict__ gb,
                            int* __restrict__ topi, float* __restrict__ topw,
                            float* __restrict__ stats) {
  int row = blockIdx.x, lane = threadIdx.x;
  __shared__ float lg[NE], pr[NE];
  if (lane < NE) {
    float acc = gb[lane];
    for (int g = 0; g < G; ++g) acc += gin[(size_t)row * G + g] * gw[(size_t)g * NE + lane];
    lg[lane] = acc;
  }
  __syncthreads();
  if (lane == 0) {
    float mx = lg[0];
    for (int e = 1; e < NE; ++e) mx = fmaxf(mx, lg[e]);
    float sum = 0.f;
    for (int e = 0; e < NE; ++e) { pr[e] = expf(lg[e] - mx); sum += pr[e]; }
    float lsum = logf(sum);
    float ent = 0.f;
    for (int e = 0; e < NE; ++e) {
      pr[e] /= sum;
      ent -= pr[e] * (lg[e] - mx - lsum);
      atomicAdd(&stats[e], pr[e]);
    }
    atomicAdd(&stats[32], ent);
    int i0 = 0; float v0 = lg[0];
    for (int e = 1; e < NE; ++e) if (lg[e] > v0) { v0 = lg[e]; i0 = e; }
    int i1 = (i0 == 0) ? 1 : 0; float v1 = lg[i1];
    for (int e = 0; e < NE; ++e) if (e != i0 && lg[e] > v1) { v1 = lg[e]; i1 = e; }
    float e0 = 1.f, e1 = expf(v1 - v0);
    float wsum = e0 + e1;
    topi[row * 2] = i0; topi[row * 2 + 1] = i1;
    topw[row * 2] = e0 / wsum; topw[row * 2 + 1] = e1 / wsum;
    atomicAdd(&stats[16 + i0], 1.f);
    atomicAdd(&stats[16 + i1], 1.f);
  }
}

// y[n] = w0*allo[i0][n] + w1*allo[i1][n]   (allo laid out (E, M, EOUT))
__global__ void combine_kernel(const float* __restrict__ allo, const int* __restrict__ topi,
                               const float* __restrict__ topw, float* __restrict__ y, int M) {
  int n = blockIdx.x, c = threadIdx.x;    // 256
  int i0 = topi[n * 2], i1 = topi[n * 2 + 1];
  float w0 = topw[n * 2], w1 = topw[n * 2 + 1];
  y[(size_t)n * EOUT + c] = w0 * allo[((size_t)i0 * M + n) * EOUT + c]
                          + w1 * allo[((size_t)i1 * M + n) * EOUT + c];
}

// ---------------- Heads ----------------
__global__ void pred_kernel(const float* __restrict__ tcn_last, const float* __restrict__ ctxm,
                            const float* __restrict__ yf, const float* __restrict__ w,
                            const float* __restrict__ bb, const float* __restrict__ last_val,
                            const float* __restrict__ mask, float* __restrict__ out) {
  int t = blockIdx.x * blockDim.x + threadIdx.x;
  if (t >= NSEQ * 3) return;
  int n = t / 3, j = t % 3, b = n >> 5;
  float acc = bb[j];
  for (int e = 0; e < TCNC; ++e) acc += tcn_last[(size_t)n * TCNC + e] * w[e * 3 + j];
  for (int e = 0; e < DM;   ++e) acc += ctxm[(size_t)n * DM + e] * w[(TCNC + e) * 3 + j];
  for (int e = 0; e < EOUT; ++e) acc += yf[(size_t)b * EOUT + e] * w[(TCNC + DM + e) * 3 + j];
  out[n * 3 + j] = (acc + last_val[n]) * mask[n];
}

__global__ void fail_kernel(const float* __restrict__ yfl, const float* __restrict__ w,
                            const float* __restrict__ bb, float* __restrict__ out) {
  int t = blockIdx.x * blockDim.x + threadIdx.x;
  if (t >= BB * 3) return;
  int b = t / 3, j = t % 3;
  float acc = bb[j];
  for (int c = 0; c < EOUT; ++c) acc += yfl[(size_t)b * EOUT + c] * w[c * 3 + j];
  out[t] = acc;
}

__global__ void rca_kernel(const float* __restrict__ tcn_last, const float* __restrict__ ctxm,
                           const float* __restrict__ yr, const float* __restrict__ w,
                           const float* __restrict__ bb, const float* __restrict__ mask,
                           float* __restrict__ out) {
  int n = blockIdx.x * blockDim.x + threadIdx.x;
  if (n >= NSEQ) return;
  float acc = bb[0];
  for (int e = 0; e < TCNC; ++e) acc += tcn_last[(size_t)n * TCNC + e] * w[e];
  for (int e = 0; e < DM;   ++e) acc += ctxm[(size_t)n * DM + e] * w[TCNC + e];
  for (int e = 0; e < EOUT; ++e) acc += yr[(size_t)n * EOUT + e] * w[TCNC + DM + e];
  out[n] = acc * mask[n];
}

__global__ void aux_kernel(const float* __restrict__ stats, float* __restrict__ out) {
  if (threadIdx.x != 0 || blockIdx.x != 0) return;
  const float ns[3] = {16.f, 16.f, 512.f};
  float lb_sum = 0.f, ent_sum = 0.f;
  for (int m = 0; m < 3; ++m) {
    const float* s = stats + m * 33;
    float n = ns[m];
    float lb = 0.f;
    for (int e = 0; e < NE; ++e) lb += (s[e] / n) * (s[16 + e] / (n * 2.f));
    lb_sum += 16.f * lb;
    ent_sum += s[32] / n;
  }
  out[0] = 0.01f * lb_sum - 0.001f * ent_sum;
}

// ---------------- Host launch ----------------
static inline void launch_gemm(const __bf16* A, int lda, unsigned long long asb,
                               const __bf16* Bt, unsigned long long bsb,
                               const float* bias, unsigned long long biassb,
                               float* C, int ldc, unsigned long long csb, __bf16* Cbf,
                               int M, int N, int K, int flags, int batch, hipStream_t st) {
  int tiles = (M / 16) * (N / 32);
  dim3 g((tiles + 3) / 4, 1, batch);
  gemm_kernel<<<g, 128, 0, st>>>(A, lda, asb, Bt, bsb, bias, biassb, C, ldc, csb, Cbf, M, N, K, flags);
}

static inline void launch_conv(const __bf16* X, const __bf16* CW, const float* bias,
                               const __bf16* R, __bf16* Y, int cin, int dil, int KT,
                               int flags, hipStream_t st) {
  int tiles = NSEQ * (TCNC / 32) * (LL / 32);
  conv_gemm_kernel<<<(tiles + 3) / 4, 128, 0, st>>>(X, CW, bias, R, Y, cin, TCNC, dil, KT, flags);
}

static inline void launch_repack(const float* W, __bf16* O, int K, int N, int batch, hipStream_t st) {
  int total = K * N;
  dim3 g((total + 255) / 256, 1, batch);
  repack_wt_kernel<<<g, 256, 0, st>>>(W, O, K, N);
}

extern "C" void kernel_launch(void* const* d_in, const int* in_sizes, int n_in,
                              void* d_out, int out_size, void* d_ws, size_t ws_size,
                              hipStream_t stream) {
  const float* const* P = (const float* const*)d_in;
  const float* x        = P[0];
  const float* mask     = P[1];
  const float* proj_w   = P[2];
  const float* proj_b   = P[3];
  // tcn level 0: 4..9 (c1w,c1b,c2w,c2b,dw,db); levels 1..3: 10+(i-1)*4
  const float* fn_g     = P[22];
  const float* fn_b     = P[23];
  const float* p2t_w    = P[24];
  const float* p2t_b    = P[25];
  const float* inter_pos= P[26];
  // layers: base 27 + 16*layer
  const float* on_g     = P[59];
  const float* on_b     = P[60];
  const float* gf_w     = P[61]; const float* gf_b = P[62];
  const float* gl_w     = P[63]; const float* gl_b = P[64];
  const float* gr_w     = P[65]; const float* gr_b = P[66];
  const float* ew1      = P[67]; const float* eb1 = P[68];
  const float* ew2      = P[69]; const float* eb2 = P[70];
  const float* pred_w   = P[71]; const float* pred_b = P[72];
  const float* fail_w   = P[73]; const float* fail_b = P[74];
  const float* rca_w    = P[75]; const float* rca_b  = P[76];

  // ---- workspace layout ----
  char* w = (char*)d_ws;
  size_t off = 0;
  auto AL = [&](size_t bytes) -> char* {
    char* p = w + off;
    off += (bytes + 255) & ~(size_t)255;
    return p;
  };
  // activations (channel-contiguous)
  __bf16* X0 = (__bf16*)AL((size_t)NSEQ * LL * PROJC * 2);
  __bf16* B0 = (__bf16*)AL((size_t)NSEQ * LL * TCNC * 2);
  __bf16* B1 = (__bf16*)AL((size_t)NSEQ * LL * TCNC * 2);
  __bf16* B2 = (__bf16*)AL((size_t)NSEQ * LL * TCNC * 2);
  // repacked conv weights bf16 [KT][cout][cin]
  __bf16* cw_l0c1 = (__bf16*)AL((size_t)3 * TCNC * PROJC * 2);
  __bf16* cw_l0dw = (__bf16*)AL((size_t)1 * TCNC * PROJC * 2);
  __bf16* cw_c2[4];
  __bf16* cw_c1[4];
  cw_c1[0] = cw_l0c1;
  for (int i = 0; i < 4; ++i) {
    if (i > 0) cw_c1[i] = (__bf16*)AL((size_t)3 * TCNC * TCNC * 2);
    cw_c2[i] = (__bf16*)AL((size_t)3 * TCNC * TCNC * 2);
  }
  // repacked dense weights bf16 (N,K)
  __bf16* p2t_t = (__bf16*)AL((size_t)TCNC * DM * 2);
  __bf16* lw_t[2][6]; // wq,wk,wv,wo,f1,f2
  for (int l = 0; l < 2; ++l) {
    for (int j = 0; j < 4; ++j) lw_t[l][j] = (__bf16*)AL((size_t)DM * DM * 2);
    lw_t[l][4] = (__bf16*)AL((size_t)DM * FFD * 2);
    lw_t[l][5] = (__bf16*)AL((size_t)FFD * DM * 2);
  }
  __bf16* ew1_t = (__bf16*)AL((size_t)NE * DM * EHID * 2);
  __bf16* ew2_t = (__bf16*)AL((size_t)NE * EHID * EOUT * 2);
  // fp32 tensors
  float* last_val = (float*)AL(NSEQ * 4);
  float* pooled   = (float*)AL((size_t)NSEQ * TCNC * 4);
  __bf16* pooled_bf = (__bf16*)AL((size_t)NSEQ * TCNC * 2);
  float* tcn_last = (float*)AL((size_t)NSEQ * TCNC * 4);
  float* z    = (float*)AL((size_t)NSEQ * DM * 4);
  float* ybuf = (float*)AL((size_t)NSEQ * DM * 4);
  __bf16* ybuf_bf = (__bf16*)AL((size_t)NSEQ * DM * 2);
  float* qb   = (float*)AL((size_t)NSEQ * DM * 4);
  float* kb   = (float*)AL((size_t)NSEQ * DM * 4);
  float* vb   = (float*)AL((size_t)NSEQ * DM * 4);
  float* ob   = (float*)AL((size_t)NSEQ * DM * 4);
  __bf16* ob_bf = (__bf16*)AL((size_t)NSEQ * DM * 2);
  float* ffb  = (float*)AL((size_t)NSEQ * FFD * 4);
  __bf16* ffb_bf = (__bf16*)AL((size_t)NSEQ * FFD * 2);
  float* ctx  = (float*)AL((size_t)NSEQ * DM * 4);
  float* ctxm = (float*)AL((size_t)NSEQ * DM * 4);
  __bf16* ctxm_bf = (__bf16*)AL((size_t)NSEQ * DM * 2);
  float* meanb = (float*)AL((size_t)BB * DM * 4);
  __bf16* meanb_bf = (__bf16*)AL((size_t)BB * DM * 2);
  float* stdb  = (float*)AL((size_t)BB * DM * 4);
  float* rin   = (float*)AL((size_t)BB * 2 * DM * 4);
  float* hs_s   = (float*)AL((size_t)NE * BB * EHID * 4);
  __bf16* hs_s_bf = (__bf16*)AL((size_t)NE * BB * EHID * 2);
  float* allo_s = (float*)AL((size_t)NE * BB * EOUT * 4);
  float* hs_t   = (float*)AL((size_t)NE * NSEQ * EHID * 4);
  __bf16* hs_t_bf = (__bf16*)AL((size_t)NE * NSEQ * EHID * 2);
  float* allo_t = (float*)AL((size_t)NE * NSEQ * EOUT * 4);
  float* yf  = (float*)AL((size_t)BB * EOUT * 4);
  float* yfl = (float*)AL((size_t)BB * EOUT * 4);
  float* yr  = (float*)AL((size_t)NSEQ * EOUT * 4);
  int*   gf_i = (int*)AL(BB * 2 * 4);   float* gf_wt = (float*)AL(BB * 2 * 4);
  int*   gl_i = (int*)AL(BB * 2 * 4);   float* gl_wt = (float*)AL(BB * 2 * 4);
  int*   gr_i = (int*)AL(NSEQ * 2 * 4); float* gr_wt = (float*)AL(NSEQ * 2 * 4);
  float* stats = (float*)AL(3 * 33 * 4);
  (void)ws_size; (void)in_sizes; (void)n_in; (void)out_size;

  hipMemsetAsync(stats, 0, 3 * 33 * 4, stream);

  // 0) repack weights (bf16, GEMM-friendly layouts)
  {
    int t0 = TCNC * PROJC * 3;
    repack_cw_kernel<<<(t0 + 255) / 256, 256, 0, stream>>>(P[4], cw_l0c1, TCNC, PROJC, 3);
    int t1 = TCNC * PROJC;
    repack_cw_kernel<<<(t1 + 255) / 256, 256, 0, stream>>>(P[8], cw_l0dw, TCNC, PROJC, 1);
    int t2 = TCNC * TCNC * 3;
    repack_cw_kernel<<<(t2 + 255) / 256, 256, 0, stream>>>(P[6], cw_c2[0], TCNC, TCNC, 3);
    for (int i = 1; i < 4; ++i) {
      repack_cw_kernel<<<(t2 + 255) / 256, 256, 0, stream>>>(P[10 + (i - 1) * 4], cw_c1[i], TCNC, TCNC, 3);
      repack_cw_kernel<<<(t2 + 255) / 256, 256, 0, stream>>>(P[12 + (i - 1) * 4], cw_c2[i], TCNC, TCNC, 3);
    }
    launch_repack(p2t_w, p2t_t, TCNC, DM, 1, stream);
    for (int l = 0; l < 2; ++l) {
      int pb0 = 27 + 16 * l;
      launch_repack(P[pb0 + 2], lw_t[l][0], DM, DM, 1, stream);   // wq
      launch_repack(P[pb0 + 4], lw_t[l][1], DM, DM, 1, stream);   // wk
      launch_repack(P[pb0 + 6], lw_t[l][2], DM, DM, 1, stream);   // wv
      launch_repack(P[pb0 + 8], lw_t[l][3], DM, DM, 1, stream);   // wo
      launch_repack(P[pb0 + 12], lw_t[l][4], DM, FFD, 1, stream); // f1w
      launch_repack(P[pb0 + 14], lw_t[l][5], FFD, DM, 1, stream); // f2w
    }
    launch_repack(ew1, ew1_t, DM, EHID, NE, stream);
    launch_repack(ew2, ew2_t, EHID, EOUT, NE, stream);
  }

  // 1) projection + PE
  proj_pe_kernel<<<NSEQ * LL, PROJC, 0, stream>>>(x, proj_w, proj_b, X0, last_val);

  // 2) TCN (conv2 fuses residual + relu)
  launch_conv(X0, cw_c1[0], P[5], nullptr, B1, PROJC, 1, 3, CF_RELU, stream);
  launch_conv(X0, cw_l0dw, P[9], nullptr, B0, PROJC, 1, 1, 0, stream);
  launch_conv(B1, cw_c2[0], P[7], B0, B2, TCNC, 1, 3, CF_RESID | CF_RELU, stream);   // h1 -> B2
  launch_conv(B2, cw_c1[1], P[11], nullptr, B0, TCNC, 2, 3, CF_RELU, stream);
  launch_conv(B0, cw_c2[1], P[13], B2, B1, TCNC, 2, 3, CF_RESID | CF_RELU, stream);  // h2 -> B1
  launch_conv(B1, cw_c1[2], P[15], nullptr, B0, TCNC, 4, 3, CF_RELU, stream);
  launch_conv(B0, cw_c2[2], P[17], B1, B2, TCNC, 4, 3, CF_RESID | CF_RELU, stream);  // h3 -> B2
  launch_conv(B2, cw_c1[3], P[19], nullptr, B0, TCNC, 8, 3, CF_RELU, stream);
  launch_conv(B0, cw_c2[3], P[21], B2, B1, TCNC, 8, 3, CF_RESID | CF_RELU, stream);  // h4 -> B1

  // 3) LN + masked pool + last tap
  ln_pool_kernel<<<NSEQ, TCNC, 0, stream>>>(B1, fn_g, fn_b, mask, pooled, pooled_bf, tcn_last);

  // 4) project to token dim, add inter_pos
  launch_gemm(pooled_bf, TCNC, 0, p2t_t, 0, p2t_b, 0, z, DM, 0, nullptr,
              NSEQ, DM, TCNC, GF_BIAS, 1, stream);
  add_pos_kernel<<<NSEQ, DM, 0, stream>>>(z, inter_pos);

  // 5) transformer layers
  for (int layer = 0; layer < 2; ++layer) {
    int pb0 = 27 + 16 * layer;
    const float *ln1g = P[pb0+0], *ln1b = P[pb0+1];
    const float *bq = P[pb0+3], *bk = P[pb0+5], *bv = P[pb0+7], *bo = P[pb0+9];
    const float *ln2g = P[pb0+10], *ln2b = P[pb0+11];
    const float *f1b = P[pb0+13], *f2b = P[pb0+15];

    ln_rows_kernel<<<NSEQ, DM, DM * 4, stream>>>(z, ln1g, ln1b, ybuf, ybuf_bf, DM);
    launch_gemm(ybuf_bf, DM, 0, lw_t[layer][0], 0, bq, 0, qb, DM, 0, nullptr, NSEQ, DM, DM, GF_BIAS, 1, stream);
    launch_gemm(ybuf_bf, DM, 0, lw_t[layer][1], 0, bk, 0, kb, DM, 0, nullptr, NSEQ, DM, DM, GF_BIAS, 1, stream);
    launch_gemm(ybuf_bf, DM, 0, lw_t[layer][2], 0, bv, 0, vb, DM, 0, nullptr, NSEQ, DM, DM, GF_BIAS, 1, stream);
    attn_kernel<<<BB * NHEAD, 32, 0, stream>>>(qb, kb, vb, mask, ob, ob_bf);
    launch_gemm(ob_bf, DM, 0, lw_t[layer][3], 0, bo, 0, z, DM, 0, nullptr,
                NSEQ, DM, DM, GF_BIAS | GF_ADD, 1, stream);
    ln_rows_kernel<<<NSEQ, DM, DM * 4, stream>>>(z, ln2g, ln2b, ybuf, ybuf_bf, DM);
    launch_gemm(ybuf_bf, DM, 0, lw_t[layer][4], 0, f1b, 0, ffb, FFD, 0, ffb_bf,
                NSEQ, FFD, DM, GF_BIAS | GF_RELU, 1, stream);
    launch_gemm(ffb_bf, FFD, 0, lw_t[layer][5], 0, f2b, 0, z, DM, 0, nullptr,
                NSEQ, DM, FFD, GF_BIAS | GF_ADD, 1, stream);
  }

  // 6) output LN + masked stats
  ln_rows_kernel<<<NSEQ, DM, DM * 4, stream>>>(z, on_g, on_b, ctx, nullptr, DM);
  ctx_stats_kernel<<<BB, DM, 0, stream>>>(ctx, mask, ctxm, ctxm_bf, meanb, meanb_bf, stdb, rin);

  // 7) MoE gating
  gate_kernel<<<BB,   32, 0, stream>>>(rin,  2 * DM, gf_w, gf_b, gf_i, gf_wt, stats + 0);
  gate_kernel<<<BB,   32, 0, stream>>>(rin,  2 * DM, gl_w, gl_b, gl_i, gl_wt, stats + 33);
  gate_kernel<<<NSEQ, 32, 0, stream>>>(ctxm, DM,     gr_w, gr_b, gr_i, gr_wt, stats + 66);

  // 8) dense all-expert compute (batched over gridDim.z = 16 experts)
  launch_gemm(meanb_bf, DM, 0, ew1_t, (unsigned long long)DM * EHID, eb1, EHID,
              hs_s, EHID, (unsigned long long)BB * EHID, hs_s_bf,
              BB, EHID, DM, GF_BIAS | GF_RELU, NE, stream);
  launch_gemm(hs_s_bf, EHID, (unsigned long long)BB * EHID, ew2_t, (unsigned long long)EHID * EOUT,
              eb2, EOUT, allo_s, EOUT, (unsigned long long)BB * EOUT, nullptr,
              BB, EOUT, EHID, GF_BIAS, NE, stream);
  launch_gemm(ctxm_bf, DM, 0, ew1_t, (unsigned long long)DM * EHID, eb1, EHID,
              hs_t, EHID, (unsigned long long)NSEQ * EHID, hs_t_bf,
              NSEQ, EHID, DM, GF_BIAS | GF_RELU, NE, stream);
  launch_gemm(hs_t_bf, EHID, (unsigned long long)NSEQ * EHID, ew2_t, (unsigned long long)EHID * EOUT,
              eb2, EOUT, allo_t, EOUT, (unsigned long long)NSEQ * EOUT, nullptr,
              NSEQ, EOUT, EHID, GF_BIAS, NE, stream);

  // 9) top-2 combine
  combine_kernel<<<BB,   EOUT, 0, stream>>>(allo_s, gf_i, gf_wt, yf,  BB);
  combine_kernel<<<BB,   EOUT, 0, stream>>>(allo_s, gl_i, gl_wt, yfl, BB);
  combine_kernel<<<NSEQ, EOUT, 0, stream>>>(allo_t, gr_i, gr_wt, yr,  NSEQ);

  // 10) heads -> d_out  [pred 1536 | fail 48 | rca 512 | aux 1]
  float* out = (float*)d_out;
  pred_kernel<<<(NSEQ * 3 + 255) / 256, 256, 0, stream>>>(tcn_last, ctxm, yf, pred_w, pred_b,
                                                          last_val, mask, out);
  fail_kernel<<<1, 64, 0, stream>>>(yfl, fail_w, fail_b, out + 1536);
  rca_kernel<<<2, 256, 0, stream>>>(tcn_last, ctxm, yr, rca_w, rca_b, mask, out + 1584);
  aux_kernel<<<1, 1, 0, stream>>>(stats, out + 2096);
}
